// MaskedMultiHeadAttention_1855425872100
// MI455X (gfx1250) — compile-verified, matrix operand order checked by reading, not run on hardware
//
#include <hip/hip_runtime.h>
#include <hip/hip_bf16.h>

typedef _Float16 half_t;
typedef __attribute__((ext_vector_type(16))) _Float16 v16h;
typedef __attribute__((ext_vector_type(8)))  _Float16 v8h;
typedef __attribute__((ext_vector_type(8)))  float    v8f;
typedef unsigned int u32x4 __attribute__((ext_vector_type(4)));
typedef int          i32x4 __attribute__((ext_vector_type(4)));
typedef int          i32x8 __attribute__((ext_vector_type(8)));

#define D_MODEL 1024
#define NHEAD   16
#define BATCH   2
#define SEQ     2048
#define DH      64
#define MROWS   (BATCH * SEQ)   // 4096

#if __has_builtin(__builtin_amdgcn_tensor_load_to_lds)
#define HAVE_TDM 1
#else
#define HAVE_TDM 0
#endif

// ---------------------------------------------------------------------------
// WMMA helper: D = A(16x32 f16) * B(32x16 f16) + C(16x16 f32)
// ---------------------------------------------------------------------------
__device__ __forceinline__ v8f wmma16(v16h a, v16h b, v8f c) {
  return __builtin_amdgcn_wmma_f32_16x16x32_f16(
      /*neg_a=*/false, a, /*neg_b=*/false, b,
      /*c_mod=*/(short)0, c, /*reuse_a=*/false, /*reuse_b=*/false);
}

__device__ __forceinline__ v8f zero8() {
  v8f z;
#pragma unroll
  for (int i = 0; i < 8; ++i) z[i] = 0.0f;
  return z;
}

// ---------------------------------------------------------------------------
// Cross-lane helpers: XOR-1/2/4 via DPP8 (VALU, no LDS pipe), XOR-8 via one
// ds_bpermute. Reductions stay inside a 16-lane half, matching the WMMA
// C-tile row distribution.
// ---------------------------------------------------------------------------
template <int MASK>
__device__ __forceinline__ float xor_dpp8(float x) {
  constexpr unsigned sel = [] {
    unsigned s = 0;
    for (int i = 0; i < 8; ++i) s |= (unsigned)((i ^ MASK) & 7) << (3 * i);
    return s;
  }();
  int r = __builtin_amdgcn_mov_dpp8(__builtin_bit_cast(int, x), sel);
  return __builtin_bit_cast(float, r);
}

__device__ __forceinline__ float reduce_max16(float v) {
  v = fmaxf(v, xor_dpp8<1>(v));
  v = fmaxf(v, xor_dpp8<2>(v));
  v = fmaxf(v, xor_dpp8<4>(v));
  v = fmaxf(v, __shfl_xor(v, 8, 32));
  return v;
}

__device__ __forceinline__ float reduce_add16(float v) {
  v += xor_dpp8<1>(v);
  v += xor_dpp8<2>(v);
  v += xor_dpp8<4>(v);
  v += __shfl_xor(v, 8, 32);
  return v;
}

// ---------------------------------------------------------------------------
// Fragment loaders. CDNA5 16-bit 16x32 operand layout (ISA 7.12.2):
//   lane L: row = L%16, K-base = (L<16 ? 0 : 8);
//   VGPR0-3 hold K = base..base+7, VGPR4-7 hold K = base+16..base+23.
// Row-major [16 rows][K contiguous] storage -> two 16-byte loads per lane.
// Generic pointer: works for both global and LDS sources.
// ---------------------------------------------------------------------------
__device__ __forceinline__ v16h load_frag_h(const half_t* p, int ld, int lane) {
  const half_t* r = p + (size_t)(lane & 15) * (size_t)ld + ((lane >> 4) << 3);
  v8h lo = *(const v8h*)(r);
  v8h hi = *(const v8h*)(r + 16);
  v16h o;
#pragma unroll
  for (int i = 0; i < 8; ++i) { o[i] = lo[i]; o[i + 8] = hi[i]; }
  return o;
}

__device__ __forceinline__ v16h load_frag_f(const float* p, int ld, int lane) {
  const float* r = p + (size_t)(lane & 15) * (size_t)ld + ((lane >> 4) << 3);
  v16h o;
#pragma unroll
  for (int i = 0; i < 8; ++i) {
    o[i]     = (_Float16)r[i];
    o[i + 8] = (_Float16)r[i + 16];
  }
  return o;
}

// ---------------------------------------------------------------------------
// Tensor Data Mover: 2D tile (tile_d1 rows of tile_d0 f16 elements, global
// row stride = stride_elems) -> LDS at lds_off, with optional per-row LDS
// padding (D# pad feature) to break bank-conflict patterns.
//   pad_interval code c -> pad after 2^(c+1) DWORDs; pad_amount code a -> a+1
//   DWORDs of padding.
// Descriptor layout per CDNA5 ISA (groups 0 and 1).
// ---------------------------------------------------------------------------
#if HAVE_TDM
__device__ __forceinline__ unsigned lds_off_of(const void* p) {
  // LDS aperture: low 32 bits of the generic address are the LDS byte offset.
  return (unsigned)(size_t)p;
}

__device__ __forceinline__ void tdm_load_2d(unsigned lds_off, const void* gsrc,
                                            unsigned tile_d0, unsigned tile_d1,
                                            unsigned long long stride_elems,
                                            unsigned tensor_d0, unsigned tensor_d1,
                                            unsigned pad_i, unsigned pad_a,
                                            bool pad_en) {
  unsigned long long ga = (unsigned long long)(size_t)gsrc;
  u32x4 g0;
  g0[0] = 1u;                                              // count=1, user D#
  g0[1] = lds_off;                                         // lds_addr
  g0[2] = (unsigned)(ga & 0xffffffffu);                    // global_addr[31:0]
  g0[3] = (unsigned)((ga >> 32) & 0x1ffffffu) | (2u << 30); // addr[56:32]|type=2
  i32x8 g1;
  g1[0] = (int)((1u << 16) |                               // data_size = 2B
                (pad_en ? (1u << 20) : 0u) |               // pad_enable
                (pad_i << 22) | (pad_a << 25));            // pad codes
  g1[1] = (int)((tensor_d0 & 0xffffu) << 16);              // tensor_dim0 lo
  g1[2] = (int)((tensor_d0 >> 16) | ((tensor_d1 & 0xffffu) << 16));
  g1[3] = (int)((tensor_d1 >> 16) | (tile_d0 << 16));      // tile_dim0
  g1[4] = (int)(tile_d1 & 0xffffu);                        // tile_dim1, dim2=0
  g1[5] = (int)(unsigned)(stride_elems & 0xffffffffull);   // dim0_stride lo
  g1[6] = (int)(unsigned)((stride_elems >> 32) & 0xffffull);
  g1[7] = 0;
  i32x4 gz = {0, 0, 0, 0};
#if __clang_major__ >= 23
  i32x8 gz8 = {0, 0, 0, 0, 0, 0, 0, 0};
  __builtin_amdgcn_tensor_load_to_lds(g0, g1, gz, gz, gz8, 0);
#else
  __builtin_amdgcn_tensor_load_to_lds(g0, g1, gz, gz, 0);
#endif
}
#endif

// ---------------------------------------------------------------------------
// Weight prep: W f32 [K][N] row-major  ->  Wt f16 [N][K] row-major.
// ---------------------------------------------------------------------------
__global__ void __launch_bounds__(256)
transpose_w_kernel(const float* __restrict__ W, half_t* __restrict__ Wt) {
  int idx = blockIdx.x * 256 + threadIdx.x;          // coalesced read
  int k = idx >> 10;
  int n = idx & (D_MODEL - 1);
  Wt[(size_t)n * D_MODEL + k] = (half_t)W[idx];
}

// ---------------------------------------------------------------------------
// 128x128-tile GEMM: out[m][n] = sum_k A[m][k] * W[k][n] + bias[n]
// Write-back modes: 0 = [B,H,S,dh] f16 ; 2 = [B,H,dh,S] f16 ; 3 = f32 [M,N].
// ---------------------------------------------------------------------------
template<bool AF16>
__global__ void __launch_bounds__(256)
__attribute__((amdgpu_waves_per_eu(2)))
gemm128_kernel(const void* __restrict__ Aptr,
               const half_t* __restrict__ Wt,
               const float* __restrict__ bias,
               half_t* __restrict__ outH,
               float* __restrict__ outF,
               int mode) {
  const int K = D_MODEL, N = D_MODEL;
  int lane = threadIdx.x & 31;
  int wave = threadIdx.x >> 5;
  int row0 = blockIdx.x * 128 + (wave >> 1) * 32;
  int col0 = blockIdx.y * 128 + (wave & 1) * 64;

  v8f acc[2][4];
#pragma unroll
  for (int mt = 0; mt < 2; ++mt)
#pragma unroll
    for (int t = 0; t < 4; ++t) acc[mt][t] = zero8();

  const float*  Af = (const float*)Aptr;
  const half_t* Ah = (const half_t*)Aptr;

  for (int k0 = 0; k0 < K; k0 += 32) {
    if (k0 + 32 < K) {   // gfx1250 global_prefetch_b8 for the next K-step
      __builtin_prefetch(Wt + (size_t)col0 * K + k0 + 32, 0, 3);
      if (AF16) __builtin_prefetch(Ah + (size_t)row0 * K + k0 + 32, 0, 3);
      else      __builtin_prefetch(Af + (size_t)row0 * K + k0 + 32, 0, 3);
    }
    v16h a0, a1;
    if (AF16) {
      a0 = load_frag_h(Ah + (size_t)row0 * K + k0, K, lane);
      a1 = load_frag_h(Ah + (size_t)(row0 + 16) * K + k0, K, lane);
    } else {
      a0 = load_frag_f(Af + (size_t)row0 * K + k0, K, lane);
      a1 = load_frag_f(Af + (size_t)(row0 + 16) * K + k0, K, lane);
    }
#pragma unroll
    for (int t = 0; t < 4; ++t) {
      v16h b = load_frag_h(Wt + (size_t)(col0 + t * 16) * K + k0, K, lane);
      acc[0][t] = wmma16(a0, b, acc[0][t]);
      acc[1][t] = wmma16(a1, b, acc[1][t]);
    }
  }

  int hsel = lane >> 4;
  int nl   = lane & 15;
#pragma unroll
  for (int mt = 0; mt < 2; ++mt) {
#pragma unroll
    for (int t = 0; t < 4; ++t) {
#pragma unroll
      for (int r = 0; r < 8; ++r) {
        int m = row0 + mt * 16 + r + hsel * 8;
        int n = col0 + t * 16 + nl;
        float val = acc[mt][t][r] + bias[n];
        if (mode == 3) {
          outF[(size_t)m * N + n] = val;
        } else {
          int b = m >> 11, s = m & (SEQ - 1);
          int h = n >> 6,  d = n & (DH - 1);
          if (mode == 0)
            outH[((size_t)(b * NHEAD + h) * SEQ + s) * DH + d] = (half_t)val;
          else
            outH[((size_t)(b * NHEAD + h) * DH + d) * SEQ + s] = (half_t)val;
        }
      }
    }
  }
}

// ---------------------------------------------------------------------------
// Causal flash attention, TDM-staged. Grid: (B*H, S/128), 8 waves/block,
// 16 queries/wave. Per 128-key superblock the workgroup cooperatively stages
// K (128 x dh) and V^T (dh x 128) tiles into LDS (TDM with per-row padding),
// then each wave streams its causal 32-key sub-blocks from LDS.
// Outer loop trip count = blockIdx.y + 1 (uniform) -> __syncthreads legal.
// Sub-block skip is wave-uniform -> EXEC all-ones around WMMA.
// ---------------------------------------------------------------------------
#define KPAD 72    // 64 + 4 DWORDs padding per row
#define VPAD 136   // 128 + 4 DWORDs padding per row

__global__ void __launch_bounds__(256)
__attribute__((amdgpu_waves_per_eu(2)))
attn_kernel(const half_t* __restrict__ Qh,
            const half_t* __restrict__ Kh,
            const half_t* __restrict__ Vt,
            half_t* __restrict__ ctx) {
  int bh   = blockIdx.x;                    // b*NHEAD + h
  int lane = threadIdx.x & 31;
  int wave = threadIdx.x >> 5;
  int q0   = blockIdx.y * 128 + wave * 16;  // this wave's query base

  const half_t* Qp = Qh + ((size_t)bh * SEQ + q0) * DH;
  const half_t* Kp = Kh + (size_t)bh * SEQ * DH;
  const half_t* Vp = Vt + (size_t)bh * DH * SEQ;

  __shared__ half_t Ks[128][KPAD];          // 18 KB, [key][dh] padded
  __shared__ half_t Vs[DH][VPAD];           // 17 KB, [dh][key] padded
  __shared__ half_t Ps[8][16][40];          // 10 KB, per-wave P tiles

  // Q fragments, resident for the whole key loop.
  v16h aq0 = load_frag_h(Qp, DH, lane);
  v16h aq1 = load_frag_h(Qp + 32, DH, lane);

  v8f o[4];
#pragma unroll
  for (int t = 0; t < 4; ++t) o[t] = zero8();
  float rmax[8], rsum[8];
#pragma unroll
  for (int r = 0; r < 8; ++r) { rmax[r] = -3.0e38f; rsum[r] = 0.0f; }

  int hsel = lane >> 4;
  int nl   = lane & 15;
  int kendW = q0 + 16;                      // this wave needs keys [0, q0+15]
  int nsb   = blockIdx.y + 1;               // 128-key superblocks (uniform)

  for (int sb = 0; sb < nsb; ++sb) {
    int kb128 = sb * 128;

    // ---- stage K/V tiles into LDS ----
#if HAVE_TDM
    if (wave == 0) {
      // K tile: 128 rows x 64 f16, contiguous rows; pad 4 DWORDs per 32 DWORDs
      tdm_load_2d(lds_off_of(&Ks[0][0]), Kp + (size_t)kb128 * DH,
                  /*tile_d0=*/DH, /*tile_d1=*/128, /*stride=*/DH,
                  /*tdim0=*/DH, /*tdim1=*/SEQ, /*pad_i=*/4, /*pad_a=*/3, true);
      __builtin_amdgcn_s_wait_tensorcnt(0);
    } else if (wave == 1) {
      // V tile: 64 rows x 128 f16, row stride SEQ; pad 4 per 64 DWORDs
      tdm_load_2d(lds_off_of(&Vs[0][0]), Vp + kb128,
                  /*tile_d0=*/128, /*tile_d1=*/DH, /*stride=*/SEQ,
                  /*tdim0=*/SEQ, /*tdim1=*/DH, /*pad_i=*/5, /*pad_a=*/3, true);
      __builtin_amdgcn_s_wait_tensorcnt(0);
    }
#else
    for (int i = threadIdx.x; i < 128 * (DH / 8); i += 256) {
      int row = i >> 3, ch = (i & 7) * 8;
      *(v8h*)&Ks[row][ch] = *(const v8h*)(Kp + (size_t)(kb128 + row) * DH + ch);
    }
    for (int i = threadIdx.x; i < DH * (128 / 8); i += 256) {
      int row = i >> 4, ch = (i & 15) * 8;
      *(v8h*)&Vs[row][ch] = *(const v8h*)(Vp + (size_t)row * SEQ + kb128 + ch);
    }
#endif
    __syncthreads();                        // tiles ready for all waves

    int kstop = kendW < kb128 + 128 ? kendW : kb128 + 128;
    for (int kb = kb128; kb < kstop; kb += 32) {   // wave-uniform trip count
      int krel = kb - kb128;

      // ---- scores: S = Q * K^T for 32 keys (two 16-key C tiles) ----
      v8f c0 = zero8(), c1 = zero8();
      c0 = wmma16(aq0, load_frag_h(&Ks[krel][0],       KPAD, lane), c0);
      c0 = wmma16(aq1, load_frag_h(&Ks[krel][32],      KPAD, lane), c0);
      c1 = wmma16(aq0, load_frag_h(&Ks[krel + 16][0],  KPAD, lane), c1);
      c1 = wmma16(aq1, load_frag_h(&Ks[krel + 16][32], KPAD, lane), c1);

      // ---- causal mask + online softmax ----
#pragma unroll
      for (int r = 0; r < 8; ++r) {
        int qi = q0 + r + hsel * 8;
        float s0 = (kb + nl      > qi) ? -1.0e30f : c0[r];
        float s1 = (kb + 16 + nl > qi) ? -1.0e30f : c1[r];
        float mb    = reduce_max16(fmaxf(s0, s1));
        float mnew  = fmaxf(rmax[r], mb);
        float scale = __expf(rmax[r] - mnew);
        float p0 = __expf(s0 - mnew);
        float p1 = __expf(s1 - mnew);
        rsum[r] = rsum[r] * scale + (p0 + p1);   // lane-local partial sum
        rmax[r] = mnew;
#pragma unroll
        for (int t = 0; t < 4; ++t) o[t][r] *= scale;

        int mrow = r + hsel * 8;                 // C-layout -> row-major LDS
        Ps[wave][mrow][nl]      = (half_t)p0;
        Ps[wave][mrow][16 + nl] = (half_t)p1;
      }

      asm volatile("s_wait_dscnt 0" ::: "memory");

      // ---- O += P * V : P as A fragment (K=32 keys), V tile in LDS ----
      v16h pf = load_frag_h(&Ps[wave][0][0], 40, lane);
#pragma unroll
      for (int t = 0; t < 4; ++t) {
        v16h bv = load_frag_h(&Vs[t * 16][krel], VPAD, lane);
        o[t] = wmma16(pf, bv, o[t]);
      }
    }
    __syncthreads();                        // all reads done before next stage
  }

  // ---- deferred row-sum reduction, normalize, store with the reference's
  //      transpose bug: flat = ((h*B + b)*S + s)*dh + d ----
  int b = bh >> 4, h = bh & 15;
#pragma unroll
  for (int r = 0; r < 8; ++r) {
    float inv = 1.0f / reduce_add16(rsum[r]);
    int s = q0 + r + hsel * 8;
    size_t base = (((size_t)h * BATCH + b) * SEQ + s) * DH;
#pragma unroll
    for (int t = 0; t < 4; ++t)
      ctx[base + t * 16 + nl] = (half_t)(o[t][r] * inv);
  }
}

// ---------------------------------------------------------------------------
// Host-side orchestration (all on `stream`, deterministic, no allocation).
// Workspace: 4x2MB Wt | 8MB Qh | 8MB Kh | 8MB Vt | 8MB ctx  (40 MB total)
// ---------------------------------------------------------------------------
extern "C" void kernel_launch(void* const* d_in, const int* in_sizes, int n_in,
                              void* d_out, int out_size, void* d_ws, size_t ws_size,
                              hipStream_t stream) {
  (void)in_sizes; (void)n_in; (void)out_size; (void)ws_size;
  const float* q   = (const float*)d_in[0];
  const float* k   = (const float*)d_in[1];
  const float* v   = (const float*)d_in[2];
  const float* w_q = (const float*)d_in[3];
  const float* b_q = (const float*)d_in[4];
  const float* w_k = (const float*)d_in[5];
  const float* b_k = (const float*)d_in[6];
  const float* w_v = (const float*)d_in[7];
  const float* b_v = (const float*)d_in[8];
  const float* w_o = (const float*)d_in[9];
  const float* b_o = (const float*)d_in[10];

  char* ws = (char*)d_ws;
  const size_t WSZ = (size_t)D_MODEL * D_MODEL * sizeof(half_t);   // 2 MB
  const size_t ASZ = (size_t)MROWS * D_MODEL * sizeof(half_t);     // 8 MB
  half_t* Wtq = (half_t*)(ws);
  half_t* Wtk = (half_t*)(ws + WSZ);
  half_t* Wtv = (half_t*)(ws + 2 * WSZ);
  half_t* Wto = (half_t*)(ws + 3 * WSZ);
  half_t* Qh  = (half_t*)(ws + 4 * WSZ);
  half_t* Kh  = (half_t*)(ws + 4 * WSZ + ASZ);
  half_t* Vt  = (half_t*)(ws + 4 * WSZ + 2 * ASZ);
  half_t* ctx = (half_t*)(ws + 4 * WSZ + 3 * ASZ);

  dim3 blk(256);

  transpose_w_kernel<<<D_MODEL * D_MODEL / 256, blk, 0, stream>>>(w_q, Wtq);
  transpose_w_kernel<<<D_MODEL * D_MODEL / 256, blk, 0, stream>>>(w_k, Wtk);
  transpose_w_kernel<<<D_MODEL * D_MODEL / 256, blk, 0, stream>>>(w_v, Wtv);
  transpose_w_kernel<<<D_MODEL * D_MODEL / 256, blk, 0, stream>>>(w_o, Wto);

  dim3 gg(MROWS / 128, D_MODEL / 128);
  gemm128_kernel<false><<<gg, blk, 0, stream>>>(q, Wtq, b_q, Qh, nullptr, 0);
  gemm128_kernel<false><<<gg, blk, 0, stream>>>(k, Wtk, b_k, Kh, nullptr, 0);
  gemm128_kernel<false><<<gg, blk, 0, stream>>>(v, Wtv, b_v, Vt, nullptr, 2);

  attn_kernel<<<dim3(BATCH * NHEAD, SEQ / 128), blk, 0, stream>>>(Qh, Kh, Vt, ctx);

  gemm128_kernel<true><<<gg, blk, 0, stream>>>(ctx, Wto, b_o, nullptr,
                                               (float*)d_out, 3);
}
